// MetaRouter_26431228740186
// MI455X (gfx1250) — matrix operand order repaired and checked by reading, not compiled, not run on hardware
//
#include <hip/hip_runtime.h>
#include <hip/hip_bf16.h>

// ---------- CDNA5 WMMA / TDM types ----------
typedef __attribute__((ext_vector_type(16))) __bf16 v16bf;
typedef __attribute__((ext_vector_type(8)))  float  v8f;
typedef unsigned __attribute__((ext_vector_type(4))) uv4;
typedef int      __attribute__((ext_vector_type(4))) iv4;
typedef int      __attribute__((ext_vector_type(8))) iv8;

union ABfrag { v16bf v; unsigned u[8]; uint4 q[2]; };

// Sizes
#define TOKD 4096
#define HID  512
#define NB   16
#define NS   2048
#define ND   16
#define NROW (NB * NS)          // 32768
#define KB_STEPS (TOKD / 32)    // 128

#if __has_builtin(__builtin_amdgcn_tensor_load_to_lds) && \
    __has_builtin(__builtin_amdgcn_s_wait_tensorcnt)
#define USE_TDM 1
#else
#define USE_TDM 0
#endif

// fp32 -> packed 2x bf16 (round-to-nearest-even)
__device__ inline unsigned pack2bf(float a, float b) {
    unsigned ua = __float_as_uint(a), ub = __float_as_uint(b);
    ua = (ua + 0x7FFFu + ((ua >> 16) & 1u)) >> 16;
    ub = (ub + 0x7FFFu + ((ub >> 16) & 1u)) >> 16;
    return (ua & 0xFFFFu) | (ub << 16);
}

__device__ inline float gelu_exact(float x) {
    return 0.5f * x * (1.0f + erff(x * 0.70710678118654752440f));
}

#if USE_TDM
// Issue one TDM load of a 32x32 fp32 tile (row stride 4096 elements) into LDS,
// padding +1 dword after each 32-dword row (LDS row stride 33 dwords).
// Descriptor per CDNA5 ISA.
__device__ inline void tdm_load_tile32x32(unsigned ldsOff, const float* gptr) {
    unsigned long long ga = (unsigned long long)(uintptr_t)gptr;
    uv4 g0;
    g0.x = 1u;                                        // count=1, user mode
    g0.y = ldsOff;                                    // lds_addr (bytes)
    g0.z = (unsigned)ga;                              // global_addr[31:0]
    g0.w = ((unsigned)(ga >> 32) & 0x01FFFFFFu) | 0x80000000u; // addr[56:32] | type=2
    iv8 g1;
    // data_size=4B (2<<16) | pad_enable (1<<20) | pad_interval=32dw (4<<22) | pad_amount=1dw (0<<25)
    g1[0] = 0x01120000;
    g1[1] = (32 << 16);   // tensor_dim0 = 32 (bits 79:48 -> low16 here)
    g1[2] = (32 << 16);   // tensor_dim1 = 32 (bits 111:80 -> low16 here)
    g1[3] = (32 << 16);   // tile_dim0 = 32 (bits 127:112)
    g1[4] = 32;           // tile_dim1 = 32 (bits 143:128), tile_dim2 = 0
    g1[5] = TOKD;         // tensor_dim0_stride = 4096 elements
    g1[6] = 0;
    g1[7] = 0;
    iv4 g2; g2[0] = 0; g2[1] = 0; g2[2] = 0; g2[3] = 0;
    iv4 g3; g3[0] = 0; g3[1] = 0; g3[2] = 0; g3[3] = 0;
#if __has_include(<hip/amd_detail/amd_gfx1250_TDM.h>)
    iv8 g4; g4[0]=0; g4[1]=0; g4[2]=0; g4[3]=0; g4[4]=0; g4[5]=0; g4[6]=0; g4[7]=0;
    __builtin_amdgcn_tensor_load_to_lds(g0, g1, g2, g3, g4, 0);
#else
    __builtin_amdgcn_tensor_load_to_lds(g0, g1, g2, g3, 0);
#endif
}
#endif

// ============================================================
// K0: pack W_proj [4096,512] fp32 -> bf16 in B-fragment layout
// Wp dword index i = ((kb*32 + nt)*32 + lane)*8 + d
// frag element j of lane L : K = kb*32 + (L>>4)*16 + j, N = nt*16 + (L&15)
// ============================================================
__global__ void pack_w_k(const float* __restrict__ W, unsigned* __restrict__ Wp) {
    int i = blockIdx.x * 256 + threadIdx.x;   // < 1048576
    int d2 = i & 7;
    int L  = (i >> 3) & 31;
    int nt = (i >> 8) & 31;
    int kb = i >> 13;
    int j0 = d2 * 2;
    int K0 = kb * 32 + (L >> 4) * 16 + j0;
    int n  = nt * 16 + (L & 15);
    float a = W[(size_t)K0 * HID + n];
    float b = W[(size_t)(K0 + 1) * HID + n];
    Wp[i] = pack2bf(a, b);
}

// ============================================================
// K1: X[32768,4096] @ Wp -> bias -> token LayerNorm -> Y[32768,512]
// block = 256 thr (8 waves = 2 m-tiles x 4 n-groups), grid = 1024
// TDM (double-buffered) stages the fp32 A tile; lanes convert to bf16
// fragments at read time; all 8 B fragments prefetched per k-step.
// ============================================================
__global__ __launch_bounds__(256) void gemm_ln_k(
    const float* __restrict__ X, const unsigned* __restrict__ Wp,
    const float* __restrict__ bp, const float* __restrict__ tg,
    const float* __restrict__ tb, float* __restrict__ Y)
{
    __shared__ float tileA[2][32 * 33];      // dbl-buf 32 rows x (32+1 pad) fp32
    __shared__ float redA[2][4][16];
    __shared__ float redB[2][4][16];

    const int tid  = threadIdx.x;
    const int w    = tid >> 5, lane = tid & 31;
    const int msub = w >> 2, ngrp = w & 3;
    const int mb   = blockIdx.x;
    const int half = lane >> 4;

    const float* Abase = X + (size_t)(mb * 32) * TOKD;

    v8f acc[8] = {};

#if USE_TDM
    if (w == 0) {
        tdm_load_tile32x32((unsigned)(uintptr_t)&tileA[0][0], Abase);
    }
#endif

    for (int kb = 0; kb < KB_STEPS; ++kb) {
        const int buf = kb & 1;
#if USE_TDM
        if (w == 0) __builtin_amdgcn_s_wait_tensorcnt(0);
        __syncthreads();
        if (w == 0 && kb + 1 < KB_STEPS) {
            tdm_load_tile32x32((unsigned)(uintptr_t)&tileA[buf ^ 1][0],
                               Abase + (size_t)(kb + 1) * 32);
        }
#else
        // fallback: thread staging of 32x32 fp32 tile into LDS
        #pragma unroll
        for (int i = tid; i < 1024; i += 256) {
            int r = i >> 5, c = i & 31;
            tileA[buf][r * 33 + c] = Abase[(size_t)r * TOKD + kb * 32 + c];
        }
        __syncthreads();
#endif
        // A fragment: row msub*16 + (lane&15), K = (lane>>4)*16 + j (matches Wp packing)
        {
            const int r = msub * 16 + (lane & 15);
            const int kbase = (lane >> 4) * 16;
            const float* tp = &tileA[buf][r * 33];
            ABfrag af;
            #pragma unroll
            for (int d2 = 0; d2 < 8; ++d2) {
                int j0 = 2 * d2;
                int K0 = kbase + j0;
                af.u[d2] = pack2bf(tp[K0], tp[K0 + 1]);
            }
            // prefetch all 8 B fragments, then issue 8 WMMAs
            ABfrag bfr[8];
            #pragma unroll
            for (int nt0 = 0; nt0 < 8; ++nt0) {
                int nt = ngrp * 8 + nt0;
                const uint4* bptr =
                    (const uint4*)(Wp + ((size_t)(kb * 32 + nt) * 32 + lane) * 8);
                bfr[nt0].q[0] = bptr[0];
                bfr[nt0].q[1] = bptr[1];
            }
            #pragma unroll
            for (int nt0 = 0; nt0 < 8; ++nt0) {
                acc[nt0] = __builtin_amdgcn_wmma_f32_16x16x32_bf16(
                    false, af.v, false, bfr[nt0].v, (short)0, acc[nt0], false, false);
            }
        }
#if !USE_TDM
        __syncthreads();
#endif
    }
#if USE_TDM
    __syncthreads();   // all waves done with tileA before LN scratch reuse phase
#endif

    // ---- epilogue: bias + LayerNorm over H=512 per row ----
    float s1[8] = {}, s2[8] = {};
    #pragma unroll
    for (int nt0 = 0; nt0 < 8; ++nt0) {
        int col = ngrp * 128 + nt0 * 16 + (lane & 15);
        float bias = bp[col];
        #pragma unroll
        for (int v = 0; v < 8; ++v) {
            float val = acc[nt0][v] + bias;
            acc[nt0][v] = val;
            s1[v] += val;
            s2[v] += val * val;
        }
    }
    #pragma unroll
    for (int v = 0; v < 8; ++v) {
        #pragma unroll
        for (int mk = 1; mk <= 8; mk <<= 1) {
            s1[v] += __shfl_xor(s1[v], mk, 32);
            s2[v] += __shfl_xor(s2[v], mk, 32);
        }
    }
    if ((lane & 15) == 0) {
        #pragma unroll
        for (int v = 0; v < 8; ++v) {
            redA[msub][ngrp][v + 8 * half] = s1[v];
            redB[msub][ngrp][v + 8 * half] = s2[v];
        }
    }
    __syncthreads();

    float mean[8], inv[8];
    #pragma unroll
    for (int v = 0; v < 8; ++v) {
        int r = v + 8 * half;
        float a = redA[msub][0][r] + redA[msub][1][r] + redA[msub][2][r] + redA[msub][3][r];
        float q = redB[msub][0][r] + redB[msub][1][r] + redB[msub][2][r] + redB[msub][3][r];
        float m_ = a * (1.0f / 512.0f);
        mean[v] = m_;
        inv[v]  = rsqrtf(q * (1.0f / 512.0f) - m_ * m_ + 1e-5f);
    }
    #pragma unroll
    for (int nt0 = 0; nt0 < 8; ++nt0) {
        int col = ngrp * 128 + nt0 * 16 + (lane & 15);
        float g = tg[col], bb = tb[col];
        #pragma unroll
        for (int v = 0; v < 8; ++v) {
            int row = mb * 32 + msub * 16 + v + 8 * half;
            Y[(size_t)row * HID + col] = (acc[nt0][v] - mean[v]) * inv[v] * g + bb;
        }
    }
}

// ============================================================
// K2: scores. 1 wave per token row; 17 dot products (global + 16 domains)
// ============================================================
__global__ __launch_bounds__(256) void scores_k(
    const float* __restrict__ x, const float* __restrict__ gq,
    const float* __restrict__ dq, const int* __restrict__ mask,
    float* __restrict__ gs, float* __restrict__ dsc)
{
    __shared__ float q[17][512];
    int tid = threadIdx.x;
    for (int i = tid; i < 17 * 512; i += 256) {
        int qq = i >> 9, h = i & 511;
        q[qq][h] = (qq == 0) ? gq[h] : dq[(size_t)(qq - 1) * HID + h];
    }
    __syncthreads();

    int w = tid >> 5, lane = tid & 31;
    int sg = blockIdx.x * 8 + w;             // global row 0..32767
    int b = sg >> 11, s = sg & 2047;
    float xr[16];
    const float* xp = x + (size_t)sg * HID;
    #pragma unroll
    for (int i = 0; i < 16; ++i) xr[i] = xp[lane + 32 * i];
    int m = mask[sg];

    for (int j = 0; j < 17; ++j) {
        float p = 0.0f;
        #pragma unroll
        for (int i = 0; i < 16; ++i) p += xr[i] * q[j][lane + 32 * i];
        #pragma unroll
        for (int mk = 16; mk >= 1; mk >>= 1) p += __shfl_xor(p, mk, 32);
        if (lane == 0) {
            float sc = (m == 0) ? -1e9f : p;
            if (j == 0) gs[sg] = sc;
            else        dsc[((size_t)(b * ND + (j - 1))) * NS + s] = sc;
        }
    }
}

// ============================================================
// K3: row softmax over S=2048 (272 rows: 16 global + 256 domain)
// ============================================================
__global__ __launch_bounds__(256) void softmax_k(float* gs, float* dsc) {
    int r = blockIdx.x;
    float* p = (r < NB) ? (gs + (size_t)r * NS) : (dsc + (size_t)(r - NB) * NS);
    int tid = threadIdx.x;
    __shared__ float red[256];

    float v[8];
    float mx = -3.4e38f;
    #pragma unroll
    for (int i = 0; i < 8; ++i) { v[i] = p[tid + 256 * i]; mx = fmaxf(mx, v[i]); }
    red[tid] = mx; __syncthreads();
    for (int off = 128; off; off >>= 1) {
        if (tid < off) red[tid] = fmaxf(red[tid], red[tid + off]);
        __syncthreads();
    }
    mx = red[0]; __syncthreads();

    float sm = 0.0f;
    #pragma unroll
    for (int i = 0; i < 8; ++i) { v[i] = expf(v[i] - mx); sm += v[i]; }
    red[tid] = sm; __syncthreads();
    for (int off = 128; off; off >>= 1) {
        if (tid < off) red[tid] += red[tid + off];
        __syncthreads();
    }
    float inv = 1.0f / red[0];
    #pragma unroll
    for (int i = 0; i < 8; ++i) p[tid + 256 * i] = v[i] * inv;
}

// ============================================================
// K4a: partial weighted context sums over s-chunks of 128
// part[b][chunk][q=0..16][h]
// ============================================================
__global__ __launch_bounds__(256) void ctx_part_k(
    const float* __restrict__ x, const float* __restrict__ gs,
    const float* __restrict__ dsc, float* __restrict__ part)
{
    int b = blockIdx.x, c = blockIdx.y;
    int tid = threadIdx.x;
    __shared__ float wts[17][128];
    int s0 = c * 128;
    for (int i = tid; i < 17 * 128; i += 256) {
        int q = i >> 7, s = i & 127;
        wts[q][s] = (q == 0) ? gs[(size_t)b * NS + s0 + s]
                             : dsc[((size_t)(b * ND + (q - 1))) * NS + s0 + s];
    }
    __syncthreads();

    int h0 = tid * 2;
    float a0[17] = {}, a1[17] = {};
    for (int s = 0; s < 128; ++s) {
        float2 xv = *(const float2*)(x + ((size_t)(b * NS + s0 + s)) * HID + h0);
        #pragma unroll
        for (int q = 0; q < 17; ++q) {
            float wv = wts[q][s];
            a0[q] += wv * xv.x;
            a1[q] += wv * xv.y;
        }
    }
    #pragma unroll
    for (int q = 0; q < 17; ++q) {
        size_t idx = (((size_t)(b * 16 + c) * 17 + q)) * HID + h0;
        part[idx]     = a0[q];
        part[idx + 1] = a1[q];
    }
}

// ============================================================
// K4b: reduce chunks + LayerNorm (gln for q=0, cln for q>0)
// ctx[b][q][h]
// ============================================================
__global__ __launch_bounds__(256) void ctx_reduce_k(
    const float* __restrict__ part, const float* __restrict__ gg,
    const float* __restrict__ gb, const float* __restrict__ cg,
    const float* __restrict__ cb, float* __restrict__ ctx)
{
    int b = blockIdx.x, tid = threadIdx.x;
    __shared__ float r1[256], r2[256];
    int h0 = tid * 2;
    for (int q = 0; q < 17; ++q) {
        float v0 = 0.0f, v1 = 0.0f;
        #pragma unroll
        for (int c = 0; c < 16; ++c) {
            size_t idx = (((size_t)(b * 16 + c) * 17 + q)) * HID + h0;
            v0 += part[idx];
            v1 += part[idx + 1];
        }
        r1[tid] = v0 + v1;
        r2[tid] = v0 * v0 + v1 * v1;
        __syncthreads();
        for (int off = 128; off; off >>= 1) {
            if (tid < off) { r1[tid] += r1[tid + off]; r2[tid] += r2[tid + off]; }
            __syncthreads();
        }
        float mean = r1[0] * (1.0f / 512.0f);
        float var  = r2[0] * (1.0f / 512.0f) - mean * mean;
        float inv  = rsqrtf(var + 1e-5f);
        const float* G  = (q == 0) ? gg : cg;
        const float* Bt = (q == 0) ? gb : cb;
        size_t o = ((size_t)b * 17 + q) * HID + h0;
        ctx[o]     = (v0 - mean) * inv * G[h0]     + Bt[h0];
        ctx[o + 1] = (v1 - mean) * inv * G[h0 + 1] + Bt[h0 + 1];
        __syncthreads();
    }
}

// ============================================================
// K5: fused FFN + gelu + LN + gelu + output head.  1 block per (b,d).
// ============================================================
__global__ __launch_bounds__(256) void ffn_out_k(
    const float* __restrict__ ctx, const float* __restrict__ W1,
    const float* __restrict__ b1, const float* __restrict__ W2,
    const float* __restrict__ b2, const float* __restrict__ fg,
    const float* __restrict__ fb, const float* __restrict__ Wo,
    const float* __restrict__ bo, const float* __restrict__ logt,
    float* __restrict__ out)
{
    int bd = blockIdx.x;
    int b = bd >> 4, d = bd & 15;
    int tid = threadIdx.x;
    __shared__ float fused[1024];
    __shared__ float h1[512];
    __shared__ float r1[256], r2[256];

    for (int i = tid; i < 512; i += 256) {
        fused[i]       = ctx[((size_t)b * 17 + 1 + d) * HID + i];   // d_ctx
        fused[512 + i] = ctx[((size_t)b * 17 + 0) * HID + i];       // g_ctx
    }
    __syncthreads();

    int c0 = tid * 2;
    float o0 = b1[c0], o1 = b1[c0 + 1];
    for (int k = 0; k < 1024; ++k) {
        float f = fused[k];
        float2 wv = *(const float2*)(W1 + (size_t)k * HID + c0);
        o0 += f * wv.x;
        o1 += f * wv.y;
    }
    o0 = gelu_exact(o0);
    o1 = gelu_exact(o1);

    // LayerNorm (fln)
    r1[tid] = o0 + o1;
    r2[tid] = o0 * o0 + o1 * o1;
    __syncthreads();
    for (int off = 128; off; off >>= 1) {
        if (tid < off) { r1[tid] += r1[tid + off]; r2[tid] += r2[tid + off]; }
        __syncthreads();
    }
    float mean = r1[0] * (1.0f / 512.0f);
    float var  = r2[0] * (1.0f / 512.0f) - mean * mean;
    __syncthreads();
    float inv = rsqrtf(var + 1e-5f);
    float n0 = (o0 - mean) * inv * fg[c0]     + fb[c0];
    float n1 = (o1 - mean) * inv * fg[c0 + 1] + fb[c0 + 1];
    h1[c0] = n0;
    h1[c0 + 1] = n1;
    __syncthreads();

    float p0 = b2[c0], p1 = b2[c0 + 1];
    for (int k = 0; k < 512; ++k) {
        float f = h1[k];
        float2 wv = *(const float2*)(W2 + (size_t)k * HID + c0);
        p0 += f * wv.x;
        p1 += f * wv.y;
    }
    p0 = gelu_exact(p0);
    p1 = gelu_exact(p1);

    r1[tid] = p0 * Wo[c0] + p1 * Wo[c0 + 1];
    __syncthreads();
    for (int off = 128; off; off >>= 1) {
        if (tid < off) r1[tid] += r1[tid + off];
        __syncthreads();
    }
    if (tid == 0) {
        float t = fminf(fmaxf(expf(logt[0]), 0.3f), 3.0f);
        out[bd] = (r1[0] + bo[0]) / t;
    }
}

// ============================================================
// Launch
// ============================================================
extern "C" void kernel_launch(void* const* d_in, const int* in_sizes, int n_in,
                              void* d_out, int out_size, void* d_ws, size_t ws_size,
                              hipStream_t stream) {
    const float* tok   = (const float*)d_in[0];
    const int*   mask  = (const int*)d_in[1];
    const float* Wproj = (const float*)d_in[2];
    const float* bproj = (const float*)d_in[3];
    const float* tlng  = (const float*)d_in[4];
    const float* tlnb  = (const float*)d_in[5];
    const float* glng  = (const float*)d_in[6];
    const float* glnb  = (const float*)d_in[7];
    const float* clng  = (const float*)d_in[8];
    const float* clnb  = (const float*)d_in[9];
    const float* flng  = (const float*)d_in[10];
    const float* flnb  = (const float*)d_in[11];
    const float* dq    = (const float*)d_in[12];
    const float* gq    = (const float*)d_in[13];
    const float* Wff1  = (const float*)d_in[14];
    const float* bff1  = (const float*)d_in[15];
    const float* Wff2  = (const float*)d_in[16];
    const float* bff2  = (const float*)d_in[17];
    const float* Wout  = (const float*)d_in[18];
    const float* bout  = (const float*)d_in[19];
    const float* logt  = (const float*)d_in[20];

    char* ws = (char*)d_ws;
    unsigned* Wp  = (unsigned*)(ws + 0);                        //  4 MB packed bf16 W
    float*    x   = (float*)(ws + ((size_t)4 << 20));           // 64 MB normalized tokens
    float*    gs  = (float*)(ws + ((size_t)68 << 20));          // 128 KB global scores
    float*    dsc = (float*)(ws + ((size_t)69 << 20));          //  2 MB domain scores
    float*    prt = (float*)(ws + ((size_t)72 << 20));          // ~8.5 MB partial ctx
    float*    ctx = (float*)(ws + ((size_t)81 << 20));          // ~0.5 MB contexts

    pack_w_k<<<4096, 256, 0, stream>>>(Wproj, Wp);
    gemm_ln_k<<<NROW / 32, 256, 0, stream>>>(tok, Wp, bproj, tlng, tlnb, x);
    scores_k<<<NROW / 8, 256, 0, stream>>>(x, gq, dq, mask, gs, dsc);
    softmax_k<<<NB + NB * ND, 256, 0, stream>>>(gs, dsc);
    ctx_part_k<<<dim3(NB, 16), 256, 0, stream>>>(x, gs, dsc, prt);
    ctx_reduce_k<<<NB, 256, 0, stream>>>(prt, glng, glnb, clng, clnb, ctx);
    ffn_out_k<<<NB * ND, 256, 0, stream>>>(ctx, Wff1, bff1, Wff2, bff2,
                                           flng, flnb, Wout, bout, logt,
                                           (float*)d_out);
}
